// RGCN_76725295775764
// MI455X (gfx1250) — matrix operand order repaired and checked by reading, not compiled, not run on hardware
//
#include <hip/hip_runtime.h>

#define NN   50000
#define EE   800000
#define RR   8
#define BBAS 4
#define DD   128
#define EPSF 1e-5f

typedef __attribute__((ext_vector_type(16))) _Float16 v16h;
typedef __attribute__((ext_vector_type(8)))  float    v8f;

// ---------------- small prep kernels ----------------

// W[r] = sum_b comp[r,b] * bases[b]   (row-major [k][f], f16)
__global__ void prep_basis_k(const float* __restrict__ comp,
                             const float* __restrict__ bases,
                             _Float16* __restrict__ W) {
  int idx = blockIdx.x * blockDim.x + threadIdx.x;   // r*16384 + k*128 + f
  if (idx >= RR * DD * DD) return;
  int r  = idx >> 14;
  int kf = idx & (DD * DD - 1);
  float acc = 0.f;
#pragma unroll
  for (int b = 0; b < BBAS; ++b)
    acc += comp[r * BBAS + b] * bases[b * DD * DD + kf];
  W[idx] = (_Float16)acc;
}

// Wc = root + skip_w (f16, row-major), biasc = bias + skip_b
__global__ void prep_comb_k(const float* __restrict__ root,
                            const float* __restrict__ skw,
                            const float* __restrict__ bias,
                            const float* __restrict__ skb,
                            _Float16* __restrict__ Wc,
                            float* __restrict__ biasc) {
  int idx = blockIdx.x * blockDim.x + threadIdx.x;
  if (idx >= DD * DD) return;
  Wc[idx] = (_Float16)(root[idx] + skw[idx]);
  if (idx < DD) biasc[idx] = bias[idx] + skb[idx];
}

__global__ void cvt_f2h_k(const float* __restrict__ in,
                          _Float16* __restrict__ out, int n) {
  int i = blockIdx.x * blockDim.x + threadIdx.x;
  if (i < n) out[i] = (_Float16)in[i];
}

// ---------------- degree / normalization ----------------

__global__ void count_k(const int* __restrict__ dst,
                        const int* __restrict__ et,
                        float* __restrict__ cnt, int E) {
  int e = blockIdx.x * blockDim.x + threadIdx.x;
  if (e < E) atomicAdd(&cnt[dst[e] * RR + et[e]], 1.0f);
}

__global__ void inv_k(float* __restrict__ cnt, int n) {
  int i = blockIdx.x * blockDim.x + threadIdx.x;
  if (i < n) cnt[i] = 1.0f / fmaxf(cnt[i], 1.0f);
}

// ---------------- WMMA GEMM: Out[M,128] = A16[M,128] @ B16[128,128] (+Cin +bias) ----------------
// Staging uses CDNA5 async global->LDS loads (ASYNCcnt) instead of VGPR round-trips.

__device__ __forceinline__ void async_g2l_b128(unsigned lds_off, const void* gaddr) {
  asm volatile("global_load_async_to_lds_b128 %0, %1, off"
               :: "v"(lds_off), "v"(gaddr) : "memory");
}

__global__ __launch_bounds__(256)
void gemm_k(const _Float16* __restrict__ A16,
            const _Float16* __restrict__ B16,   // row-major [k][f]
            float* __restrict__ Out,
            const float* __restrict__ Cin,      // optional
            const float* __restrict__ bias,     // optional
            int M) {
  // padded stride 136 halves (272B row): 16B-aligned rows, spread banks
  __shared__ _Float16 lA[128 * 136];
  __shared__ _Float16 lB[128 * 136];

  const int tid      = threadIdx.x;
  const int rowBlock = blockIdx.x * 128;

  // generic address of a shared var: addr[31:0] is the LDS byte offset (aperture rule)
  const unsigned ldsA_base = (unsigned)(size_t)(void*)lA;
  const unsigned ldsB_base = (unsigned)(size_t)(void*)lB;

  {
    int line = tid >> 1;              // 0..127
    int part = (tid & 1) * 64;        // half offset within line
    // B tile (always in range): 8 async b128 transfers per thread
    {
      const char* gB    = (const char*)(B16 + line * 128 + part);
      unsigned    lBoff = ldsB_base + (unsigned)(line * 136 + part) * 2u;
#pragma unroll
      for (int i = 0; i < 8; ++i)
        async_g2l_b128(lBoff + 16u * i, gB + 16 * i);
    }
    // A tile: async when in range, synchronous zero-fill otherwise
    int gRow = rowBlock + line;
    if (gRow < M) {
      const char* gA    = (const char*)(A16 + (size_t)gRow * 128 + part);
      unsigned    lAoff = ldsA_base + (unsigned)(line * 136 + part) * 2u;
#pragma unroll
      for (int i = 0; i < 8; ++i)
        async_g2l_b128(lAoff + 16u * i, gA + 16 * i);
    } else {
      uint4  z = {0u, 0u, 0u, 0u};
      uint4* d = (uint4*)(lA + line * 136 + part);
#pragma unroll
      for (int i = 0; i < 8; ++i) d[i] = z;
    }
  }
  asm volatile("s_wait_asynccnt 0" ::: "memory");
  __syncthreads();

  const int wave    = tid >> 5;
  const int lane    = tid & 31;
  const int rowBase = wave * 16;        // 16 rows per wave
  const int m       = lane & 15;
  const int kh      = (lane >> 4) << 3; // 0 or 8 (A layout half-split)

  v8f acc[8];
#pragma unroll
  for (int t = 0; t < 8; ++t) {
    v8f z = {0.f, 0.f, 0.f, 0.f, 0.f, 0.f, 0.f, 0.f};
    acc[t] = z;
  }

  union Frag { v16h h; unsigned int d[8]; };

  const _Float16* Arow = lA + (rowBase + m) * 136;

#pragma unroll
  for (int kk = 0; kk < 128; kk += 32) {
    // A fragment (16-bit A 16x32 layout: lane=M, VGPR0-3 K=kh+{0..7}, VGPR4-7 K=16+kh+{0..7})
    Frag a;
    const unsigned int* ap0 = (const unsigned int*)(Arow + kk + kh);
    const unsigned int* ap1 = (const unsigned int*)(Arow + kk + 16 + kh);
#pragma unroll
    for (int j = 0; j < 4; ++j) { a.d[j] = ap0[j]; a.d[4 + j] = ap1[j]; }

    // B fragments: lane = N col (lane&15), elements K=kh+{0..7}, 16+kh+{0..7} (same as A).
    // Load all 8 fragments first so the 8 WMMAs can issue back-to-back.
    Frag b[8];
    const _Float16* Bcol = lB + (kk + kh) * 136 + m;
#pragma unroll
    for (int t = 0; t < 8; ++t) {
#pragma unroll
      for (int s = 0; s < 16; ++s) b[t].h[s] = Bcol[(s < 8 ? s : 8 + s) * 136 + t * 16];
    }
#pragma unroll
    for (int t = 0; t < 8; ++t)
      acc[t] = __builtin_amdgcn_wmma_f32_16x16x32_f16(
          false, a.h, false, b[t].h, (short)0, acc[t], false, false);
  }

  // epilogue: C/D layout — lane(0-15)=N col, VGPR v = row v (lane<16) / 8+v (lane>=16)
  const int rOff = (lane >> 4) << 3;
  const int col0 = lane & 15;
#pragma unroll
  for (int t = 0; t < 8; ++t) {
    int col = t * 16 + col0;
    float bv = bias ? bias[col] : 0.f;
#pragma unroll
    for (int v = 0; v < 8; ++v) {
      int gRow = rowBlock + rowBase + rOff + v;
      if (gRow < M) {
        float val = acc[t][v] + bv;
        if (Cin) val += Cin[(size_t)gRow * 128 + col];
        Out[(size_t)gRow * 128 + col] = val;
      }
    }
  }
}

// ---------------- edge scatter (one wave32 per edge) ----------------

__global__ void scatter_k(const float* __restrict__ h,
                          const int* __restrict__ src,
                          const int* __restrict__ dst,
                          const int* __restrict__ et,
                          const float* __restrict__ inv,
                          float* __restrict__ agg, int E, int r) {
  int gid  = blockIdx.x * blockDim.x + threadIdx.x;
  int e    = gid >> 5;
  int lane = gid & 31;
  if (e >= E) return;
  if (et[e] != r) return;
  int s = src[e], d = dst[e];
  float scale = inv[d * RR + r];
  const float4* hp = (const float4*)(h + (size_t)s * 128);
  float4 v = hp[lane];
  float* ap = agg + (size_t)d * 128 + lane * 4;
  atomicAdd(ap + 0, v.x * scale);
  atomicAdd(ap + 1, v.y * scale);
  atomicAdd(ap + 2, v.z * scale);
  atomicAdd(ap + 3, v.w * scale);
}

// ---------------- BatchNorm ----------------

__global__ void bn_partial_k(const float* __restrict__ x1,
                             float* __restrict__ gsum,
                             float* __restrict__ gsq, int n) {
  int c  = threadIdx.x;                 // 128 threads = channel
  int r0 = blockIdx.x * 512;
  int r1 = min(r0 + 512, n);
  float s = 0.f, q = 0.f;
  for (int r = r0; r < r1; ++r) {
    float v = x1[(size_t)r * DD + c];
    s += v; q += v * v;
  }
  atomicAdd(&gsum[c], s);
  atomicAdd(&gsq[c], q);
}

__global__ void bn_final_k(const float* gsum, const float* gsq,
                           const float* gamma, const float* beta,
                           float* ab) {
  int c = threadIdx.x;
  if (c >= DD) return;
  float mean = gsum[c] / (float)NN;
  float var  = gsq[c] / (float)NN - mean * mean;
  float a    = gamma[c] * rsqrtf(var + EPSF);
  ab[c]      = a;
  ab[DD + c] = beta[c] - mean * a;
}

__global__ void bn_relu_cvt_k(const float* __restrict__ x1,
                              const float* __restrict__ ab,
                              _Float16* __restrict__ xr16, int n) {
  int i = blockIdx.x * blockDim.x + threadIdx.x;
  if (i >= n) return;
  int c = i & (DD - 1);
  float v = fmaxf(ab[c] * x1[i] + ab[DD + c], 0.f);
  xr16[i] = (_Float16)v;
}

// ---------------- host orchestration ----------------

extern "C" void kernel_launch(void* const* d_in, const int* in_sizes, int n_in,
                              void* d_out, int out_size, void* d_ws, size_t ws_size,
                              hipStream_t stream) {
  const float* x      = (const float*)d_in[0];
  const int*   eidx   = (const int*)  d_in[1];
  const int*   etype  = (const int*)  d_in[2];
  const float* comp1  = (const float*)d_in[3];
  const float* bases1 = (const float*)d_in[4];
  const float* root1  = (const float*)d_in[5];
  const float* bias1  = (const float*)d_in[6];
  const float* skip1w = (const float*)d_in[7];
  const float* skip1b = (const float*)d_in[8];
  const float* gamma  = (const float*)d_in[9];
  const float* beta   = (const float*)d_in[10];
  const float* comp2  = (const float*)d_in[11];
  const float* bases2 = (const float*)d_in[12];
  const float* root2  = (const float*)d_in[13];
  const float* bias2  = (const float*)d_in[14];
  const float* skip2w = (const float*)d_in[15];
  const float* skip2b = (const float*)d_in[16];

  const int* srcp = eidx;
  const int* dstp = eidx + EE;

  float* out = (float*)d_out;
  float* x1o = out;                       // h (pre-BN x1)  [N,128]
  float* x2o = out + (size_t)NN * DD;     // x2             [N,128]

  // workspace carve (all 256B aligned)
  char* wsb = (char*)d_ws;
  size_t off = 0;
  auto carve = [&](size_t bytes) -> char* {
    char* p = wsb + off;
    off = (off + bytes + 255) & ~(size_t)255;
    return p;
  };
  float*    inv   = (float*)   carve((size_t)NN * RR * 4);
  _Float16* W     = (_Float16*)carve((size_t)RR * DD * DD * 2);
  _Float16* Wc    = (_Float16*)carve((size_t)DD * DD * 2);
  float*    biasc = (float*)   carve(DD * 4);
  _Float16* x16   = (_Float16*)carve((size_t)NN * DD * 2);   // x (layer1) then xr (layer2)
  float*    hbuf  = (float*)   carve((size_t)NN * DD * 4);   // h_r, L2-resident
  float*    agg   = (float*)   carve((size_t)NN * DD * 4);   // aggregated messages
  float*    gsum  = (float*)   carve(DD * 4);
  float*    gsq   = (float*)   carve(DD * 4);
  float*    ab    = (float*)   carve(2 * DD * 4);

  const int gemmGrid = (NN + 127) / 128;
  const int nElem    = NN * DD;

  // per-(dst,rel) mean normalization factors
  hipMemsetAsync(inv, 0, (size_t)NN * RR * 4, stream);
  count_k<<<(EE + 255) / 256, 256, 0, stream>>>(dstp, etype, inv, EE);
  inv_k<<<(NN * RR + 255) / 256, 256, 0, stream>>>(inv, NN * RR);

  // ---- layer 1 ----
  prep_basis_k<<<(RR * DD * DD) / 256, 256, 0, stream>>>(comp1, bases1, W);
  prep_comb_k<<<(DD * DD) / 256, 256, 0, stream>>>(root1, skip1w, bias1, skip1b, Wc, biasc);
  cvt_f2h_k<<<(nElem + 255) / 256, 256, 0, stream>>>(x, x16, nElem);
  hipMemsetAsync(agg, 0, (size_t)NN * DD * 4, stream);
  for (int r = 0; r < RR; ++r) {
    gemm_k<<<gemmGrid, 256, 0, stream>>>(x16, W + r * DD * DD, hbuf, nullptr, nullptr, NN);
    scatter_k<<<(EE * 32 + 255) / 256, 256, 0, stream>>>(hbuf, srcp, dstp, etype, inv, agg, EE, r);
  }
  gemm_k<<<gemmGrid, 256, 0, stream>>>(x16, Wc, x1o, agg, biasc, NN);

  // ---- BatchNorm + ReLU (training-style batch stats) ----
  hipMemsetAsync(gsum, 0, DD * 4, stream);
  hipMemsetAsync(gsq,  0, DD * 4, stream);
  bn_partial_k<<<(NN + 511) / 512, DD, 0, stream>>>(x1o, gsum, gsq, NN);
  bn_final_k<<<1, DD, 0, stream>>>(gsum, gsq, gamma, beta, ab);
  bn_relu_cvt_k<<<(nElem + 255) / 256, 256, 0, stream>>>(x1o, ab, x16, nElem);

  // ---- layer 2 ----
  prep_basis_k<<<(RR * DD * DD) / 256, 256, 0, stream>>>(comp2, bases2, W);
  prep_comb_k<<<(DD * DD) / 256, 256, 0, stream>>>(root2, skip2w, bias2, skip2b, Wc, biasc);
  hipMemsetAsync(agg, 0, (size_t)NN * DD * 4, stream);
  for (int r = 0; r < RR; ++r) {
    gemm_k<<<gemmGrid, 256, 0, stream>>>(x16, W + r * DD * DD, hbuf, nullptr, nullptr, NN);
    scatter_k<<<(EE * 32 + 255) / 256, 256, 0, stream>>>(hbuf, srcp, dstp, etype, inv, agg, EE, r);
  }
  gemm_k<<<gemmGrid, 256, 0, stream>>>(x16, Wc, x2o, agg, biasc, NN);
}
